// PhyloAttention_34419867910741
// MI455X (gfx1250) — matrix operand order repaired and checked by reading, not compiled, not run on hardware
//
#include <hip/hip_runtime.h>
#include <hip/hip_bf16.h>

// ---------------- problem constants (match reference) ----------------
#define BATCH   2
#define SEQ     2048
#define NHEAD   16
#define HDIM    64
#define EMBED   1024          // NHEAD*HDIM
#define QKVN    3072          // 3*EMBED
#define SCALE_F 0.125f        // HDIM^-0.5

typedef __bf16 bf16_t;
typedef __attribute__((ext_vector_type(16))) __bf16 v16bf;
typedef __attribute__((ext_vector_type(8)))  __bf16 v8bf;
typedef __attribute__((ext_vector_type(8)))  float  v8f;

union FragAB { v16bf v; v8bf h[2]; };

__device__ __forceinline__ void lds_fence() {
    // intra-wave ordering of DS stores vs. subsequent DS loads (cross-lane
    // data exchange through LDS); also a compiler memory barrier.
    asm volatile("s_wait_dscnt 0" ::: "memory");
}

__device__ __forceinline__ v8f wmma_bf16(const v16bf& a, const v16bf& b, const v8f& c) {
    return __builtin_amdgcn_wmma_f32_16x16x32_bf16(false, a, false, b, (short)0, c, false, false);
}

// ---------------------------------------------------------------------
// Kernel 1: per-batch phylo temperature scalar
// ---------------------------------------------------------------------
__global__ __launch_bounds__(256)
void phylo_temp_kernel(const float* __restrict__ pd, const float* __restrict__ alpha,
                       float* __restrict__ temp) {
    __shared__ float red[256];
    int b = blockIdx.x;
    float s = 0.f;
    for (int i = threadIdx.x; i < SEQ; i += 256) s += pd[b * SEQ + i];
    red[threadIdx.x] = s;
    __syncthreads();
    for (int off = 128; off > 0; off >>= 1) {
        if (threadIdx.x < off) red[threadIdx.x] += red[threadIdx.x + off];
        __syncthreads();
    }
    if (threadIdx.x == 0) {
        float mean = red[0] * (1.0f / (float)SEQ);
        float t = 1.0f + alpha[0] * mean;
        temp[b] = t < 1e-6f ? 1e-6f : t;
    }
}

// ---------------------------------------------------------------------
// Kernel 2/4: bf16-WMMA GEMM  C[M,N] = A[M,K] * W[N,K]^T + bias
//   mode 0: write f32 row-major to outF (out projection)
//   mode 1: scatter bf16 Q/K (head-major [b,h,l,d]) and V^T ([b,h,d,l])
// block = 128 threads (4 waves), block tile 64x64, K-step 32
// ---------------------------------------------------------------------
__global__ __launch_bounds__(128)
void gemm_bf16_wmma(const float* __restrict__ A, const float* __restrict__ W,
                    const float* __restrict__ bias, int Nn, int Kk, int mode,
                    float* __restrict__ outF,
                    bf16_t* __restrict__ Qb, bf16_t* __restrict__ Kb, bf16_t* __restrict__ Vb) {
    __shared__ __align__(16) bf16_t a_lds[64 * 40];   // 64 rows, stride 32+8
    __shared__ __align__(16) bf16_t w_lds[64 * 40];

    const int t    = threadIdx.x;
    const int lane = t & 31;
    const int wave = t >> 5;
    const int rw   = wave >> 1;          // wave row (0..1) -> 32-row strip
    const int cw   = wave & 1;           // wave col (0..1) -> 32-col strip
    const int half = lane >> 4;          // lane group g
    const int l15  = lane & 15;

    const int bm = blockIdx.y * 64;
    const int bn = blockIdx.x * 64;

    v8f acc[2][2] = {};

    const int r0 = t >> 3;               // 0..15
    const int cb = (t & 7) * 4;          // 0,4,...,28

    for (int kb = 0; kb < Kk; kb += 32) {
        __syncthreads();                 // WAR vs previous iteration's reads
        #pragma unroll
        for (int rr = 0; rr < 4; ++rr) {
            int row = r0 + rr * 16;
            float4 fa = *(const float4*)&A[(size_t)(bm + row) * Kk + kb + cb];
            a_lds[row * 40 + cb + 0] = (bf16_t)fa.x;
            a_lds[row * 40 + cb + 1] = (bf16_t)fa.y;
            a_lds[row * 40 + cb + 2] = (bf16_t)fa.z;
            a_lds[row * 40 + cb + 3] = (bf16_t)fa.w;
            float4 fw = *(const float4*)&W[(size_t)(bn + row) * Kk + kb + cb];
            w_lds[row * 40 + cb + 0] = (bf16_t)fw.x;
            w_lds[row * 40 + cb + 1] = (bf16_t)fw.y;
            w_lds[row * 40 + cb + 2] = (bf16_t)fw.z;
            w_lds[row * 40 + cb + 3] = (bf16_t)fw.w;
        }
        __syncthreads();

        // A fragments: lane holds row m=l15; dwords 0-3: K=g*8+0..7, 4-7: K=16+g*8+0..7
        FragAB af[2];
        #pragma unroll
        for (int mi = 0; mi < 2; ++mi) {
            int row = rw * 32 + mi * 16 + l15;
            af[mi].h[0] = *(const v8bf*)&a_lds[row * 40 + half * 8];
            af[mi].h[1] = *(const v8bf*)&a_lds[row * 40 + 16 + half * 8];
        }
        // B fragments ([N][K] layout = W natural): lane holds col n; dwords 0-3: K=g*8+0..7, 4-7: K=16+g*8+0..7
        FragAB wf[2];
        #pragma unroll
        for (int ni = 0; ni < 2; ++ni) {
            int n = cw * 32 + ni * 16 + l15;
            wf[ni].h[0] = *(const v8bf*)&w_lds[n * 40 + half * 8];
            wf[ni].h[1] = *(const v8bf*)&w_lds[n * 40 + 16 + half * 8];
        }
        #pragma unroll
        for (int mi = 0; mi < 2; ++mi)
            #pragma unroll
            for (int ni = 0; ni < 2; ++ni)
                acc[mi][ni] = wmma_bf16(af[mi].v, wf[ni].v, acc[mi][ni]);
    }

    // epilogue; C layout: lane group g holds rows g*8+i, col = l15 within tile
    #pragma unroll
    for (int mi = 0; mi < 2; ++mi) {
        #pragma unroll
        for (int ni = 0; ni < 2; ++ni) {
            int colg = bn + cw * 32 + ni * 16 + l15;
            float bv = bias[colg];
            if (mode == 0) {
                #pragma unroll
                for (int i = 0; i < 8; ++i) {
                    int rowg = bm + rw * 32 + mi * 16 + half * 8 + i;
                    outF[(size_t)rowg * Nn + colg] = acc[mi][ni][i] + bv;
                }
            } else {
                int which = colg >> 10;            // 0=q 1=k 2=v
                int r     = colg & 1023;
                int h     = r >> 6;
                int d     = r & 63;
                #pragma unroll
                for (int i = 0; i < 8; ++i) {
                    int rowg = bm + rw * 32 + mi * 16 + half * 8 + i;
                    int b = rowg >> 11;            // / SEQ
                    int l = rowg & (SEQ - 1);
                    bf16_t v = (bf16_t)(acc[mi][ni][i] + bv);
                    size_t headbase = ((size_t)b * NHEAD + h);
                    if (which == 0)      Qb[(headbase * SEQ + l) * HDIM + d] = v;
                    else if (which == 1) Kb[(headbase * SEQ + l) * HDIM + d] = v;
                    else                 Vb[(headbase * HDIM + d) * SEQ + l] = v; // V^T
                }
            }
        }
    }
}

// ---------------------------------------------------------------------
// Kernel 3: flash attention, 1 wave per (b,h, 16-row q tile)
// kv tiles of 32; QK^T: 2 WMMAs per 16-col N-tile; PV: 4 WMMAs
// ---------------------------------------------------------------------
__global__ __launch_bounds__(32)
void phylo_flash_attn(const bf16_t* __restrict__ Qb, const bf16_t* __restrict__ Kb,
                      const bf16_t* __restrict__ Vb, const float* __restrict__ alibi,
                      const float* __restrict__ temp, float* __restrict__ attn_out) {
    __shared__ __align__(16) bf16_t q_lds [16 * 72];   // [q][d]   stride 64+8
    __shared__ __align__(16) bf16_t k_lds [32 * 72];   // [kv][d]  stride 64+8
    __shared__ __align__(16) bf16_t vt_lds[64 * 40];   // [d][kv]  stride 32+8
    __shared__ __align__(16) bf16_t p_lds [16 * 40];   // [q][kv]  stride 32+8

    const int lane  = threadIdx.x;
    const int half  = lane >> 4;
    const int l15   = lane & 15;
    const int qtile = blockIdx.x;
    const int bh    = blockIdx.y;
    const int b     = bh >> 4;
    const int h     = bh & 15;
    const int qbase = qtile * 16;
    const int qend  = qbase + 15;

    const float tscale = SCALE_F * temp[b];

    // ---- load 16x64 Q tile (bf16, head-major) ----
    const bf16_t* qg = Qb + ((size_t)bh * SEQ + qbase) * HDIM;
    {
        int row = l15;
        #pragma unroll
        for (int c = 0; c < 4; ++c) {
            int e = half * 32 + c * 8;
            *(v8bf*)&q_lds[row * 72 + e] = *(const v8bf*)&qg[row * HDIM + e];
        }
    }
    lds_fence();

    // Q A-fragments for the two 32-wide K steps of hd=64
    FragAB qf[2];
    #pragma unroll
    for (int ks = 0; ks < 2; ++ks) {
        qf[ks].h[0] = *(const v8bf*)&q_lds[l15 * 72 + ks * 32 + half * 8];
        qf[ks].h[1] = *(const v8bf*)&q_lds[l15 * 72 + ks * 32 + 16 + half * 8];
    }

    float m_r[8], l_r[8];
    #pragma unroll
    for (int i = 0; i < 8; ++i) { m_r[i] = -1e30f; l_r[i] = 0.f; }
    v8f acc[4] = {};

    const bf16_t* kg_base = Kb + (size_t)bh * SEQ * HDIM;
    const bf16_t* vg_base = Vb + (size_t)bh * HDIM * SEQ;   // [d][l]

    for (int kv = 0; kv <= qend; kv += 32) {
        // ---- load 32x64 K tile ----
        const bf16_t* kg = kg_base + (size_t)kv * HDIM;
        #pragma unroll
        for (int c = 0; c < 8; ++c)
            *(v8bf*)&k_lds[lane * 72 + c * 8] = *(const v8bf*)&kg[lane * HDIM + c * 8];
        // ---- load 64x32 V^T tile (V stored d-major) ----
        #pragma unroll
        for (int dr = 0; dr < 2; ++dr) {
            int d = lane + dr * 32;
            #pragma unroll
            for (int c = 0; c < 4; ++c)
                *(v8bf*)&vt_lds[d * 40 + c * 8] =
                    *(const v8bf*)&vg_base[(size_t)d * SEQ + kv + c * 8];
        }
        // hint: prefetch next K tile
        if (kv + 32 <= qend) __builtin_prefetch(kg_base + (size_t)(kv + 32) * HDIM, 0, 0);
        lds_fence();

        // ---- S = Q K^T (two 16-col N-tiles), + alibi, mask, scale ----
        float sx[2][8];
        #pragma unroll
        for (int nt = 0; nt < 2; ++nt) {
            FragAB kf[2];
            #pragma unroll
            for (int ks = 0; ks < 2; ++ks) {
                int r = (nt * 16 + l15) * 72 + ks * 32 + half * 8;
                kf[ks].h[0] = *(const v8bf*)&k_lds[r];
                kf[ks].h[1] = *(const v8bf*)&k_lds[r + 16];
            }
            v8f c = {};
            c = wmma_bf16(qf[0].v, kf[0].v, c);
            c = wmma_bf16(qf[1].v, kf[1].v, c);
            int jg = kv + nt * 16 + l15;
            #pragma unroll
            for (int i = 0; i < 8; ++i) {
                int ig = qbase + half * 8 + i;
                float sv = c[i] * tscale + alibi[((size_t)h * SEQ + ig) * SEQ + jg];
                if (jg > ig) sv = -1e30f;          // causal mask
                sx[nt][i] = sv;
            }
        }

        // ---- online softmax (row reductions across the 16-lane group) ----
        #pragma unroll
        for (int i = 0; i < 8; ++i) {
            float mx = fmaxf(sx[0][i], sx[1][i]);
            #pragma unroll
            for (int off = 1; off < 16; off <<= 1)
                mx = fmaxf(mx, __shfl_xor(mx, off, 32));
            float mnew  = fmaxf(m_r[i], mx);
            float alpha = __expf(m_r[i] - mnew);
            float p0 = __expf(sx[0][i] - mnew);
            float p1 = __expf(sx[1][i] - mnew);
            float rs = p0 + p1;
            #pragma unroll
            for (int off = 1; off < 16; off <<= 1)
                rs += __shfl_xor(rs, off, 32);
            l_r[i] = l_r[i] * alpha + rs;
            m_r[i] = mnew;
            #pragma unroll
            for (int j = 0; j < 4; ++j) acc[j][i] *= alpha;
            int row = half * 8 + i;
            p_lds[row * 40 + l15]      = (bf16_t)p0;
            p_lds[row * 40 + 16 + l15] = (bf16_t)p1;
        }
        lds_fence();

        // ---- O += P V : A = P[16x32], B = V[32xd16] via V^T LDS ----
        FragAB pf;
        pf.h[0] = *(const v8bf*)&p_lds[l15 * 40 + half * 8];
        pf.h[1] = *(const v8bf*)&p_lds[l15 * 40 + 16 + half * 8];
        #pragma unroll
        for (int j = 0; j < 4; ++j) {
            FragAB vf;
            int r = (j * 16 + l15) * 40 + half * 8;
            vf.h[0] = *(const v8bf*)&vt_lds[r];
            vf.h[1] = *(const v8bf*)&vt_lds[r + 16];
            acc[j] = wmma_bf16(pf.v, vf.v, acc[j]);
        }
        lds_fence();   // protect k_lds/vt_lds reuse next iteration
    }

    // ---- normalize and write [b, l, h*64+d] f32 ----
    #pragma unroll
    for (int i = 0; i < 8; ++i) {
        float inv = 1.0f / l_r[i];
        int rowg = qbase + half * 8 + i;
        size_t base = ((size_t)(b * SEQ + rowg)) * EMBED + h * HDIM;
        #pragma unroll
        for (int j = 0; j < 4; ++j)
            attn_out[base + j * 16 + l15] = acc[j][i] * inv;
    }
}

// ---------------------------------------------------------------------
extern "C" void kernel_launch(void* const* d_in, const int* in_sizes, int n_in,
                              void* d_out, int out_size, void* d_ws, size_t ws_size,
                              hipStream_t stream) {
    const float* x      = (const float*)d_in[0];   // [B,L,D]
    const float* phylo  = (const float*)d_in[1];   // [B,L]
    const float* alibi  = (const float*)d_in[2];   // [H,L,L]
    /* d_in[3] = attn_mask (causal, recomputed analytically) */
    const float* qkv_w  = (const float*)d_in[4];   // [3D,D]
    const float* qkv_b  = (const float*)d_in[5];   // [3D]
    const float* out_w  = (const float*)d_in[6];   // [D,D]
    const float* out_b  = (const float*)d_in[7];   // [D]
    const float* alpha  = (const float*)d_in[8];   // scalar

    // workspace layout
    char*  ws   = (char*)d_ws;
    size_t off  = 0;
    float* temp = (float*)(ws + off);  off += 256;
    const size_t headElems = (size_t)BATCH * NHEAD * SEQ * HDIM;   // 4M elems
    bf16_t* Qb = (bf16_t*)(ws + off);  off += headElems * sizeof(bf16_t);
    bf16_t* Kb = (bf16_t*)(ws + off);  off += headElems * sizeof(bf16_t);
    bf16_t* Vb = (bf16_t*)(ws + off);  off += headElems * sizeof(bf16_t);
    float* attn_out = (float*)(ws + off);  off += (size_t)BATCH * SEQ * EMBED * sizeof(float);

    // 1) phylo temperature scalars
    phylo_temp_kernel<<<BATCH, 256, 0, stream>>>(phylo, alpha, temp);

    // 2) QKV projection: [4096,1024] x [1024,3072]^T, scatter q/k/v bf16
    gemm_bf16_wmma<<<dim3(QKVN / 64, (BATCH * SEQ) / 64), 128, 0, stream>>>(
        x, qkv_w, qkv_b, QKVN, EMBED, 1, nullptr, Qb, Kb, Vb);

    // 3) flash attention: one wave per (bh, 16-row q tile)
    phylo_flash_attn<<<dim3(SEQ / 16, BATCH * NHEAD), 32, 0, stream>>>(
        Qb, Kb, Vb, alibi, temp, attn_out);

    // 4) output projection into d_out (f32)
    gemm_bf16_wmma<<<dim3(EMBED / 64, (BATCH * SEQ) / 64), 128, 0, stream>>>(
        attn_out, out_w, out_b, EMBED, EMBED, 0, (float*)d_out,
        nullptr, nullptr, nullptr);
}
